// LearnableEMA_12429635354985
// MI455X (gfx1250) — compile-verified, matrix operand order checked by reading, not run on hardware
//
#include <hip/hip_runtime.h>
#include <math.h>

// ---------------------------------------------------------------------------
// Learnable EMA scan, (B=512, T=32768) f32, y[t] = a*x[t] + (1-a)*y[t-1],
// init state = x[:,0].  Chunked linear-recurrence decomposition:
//   K1: per-chunk geometric-weighted sums (A_chunk)     [warms L2 with x]
//   K2: per-row carry chain across chunks (s' = A + c^L * s)
//   K3: full scan per chunk; per 256-element tile the 16 within-segment
//       scans are ONE v_wmma_f32_16x16x32_f16 with a lower-triangular
//       c-power matrix; segment carries via ds_bpermute scan.  Loads are
//       nontemporal (last use of x), stores nontemporal (y never re-read),
//       so x stays resident in the 192MB L2 between K1 and K3.
// ---------------------------------------------------------------------------

typedef __attribute__((ext_vector_type(16))) _Float16 v16h;
typedef __attribute__((ext_vector_type(8)))  float    v8f;
typedef __attribute__((ext_vector_type(4)))  float    v4f;   // native vec for NT ld/st

#define B_ROWS 512
#define T_LEN  32768
#define LCHUNK 2048
#define NCHUNK (T_LEN / LCHUNK)   // 16 chunks per row -> 8192 waves total
#define TILE   256
#define NTILE  (LCHUNK / TILE)    // 8 WMMA tiles per chunk

static __device__ __forceinline__ float shfl_f(float v, int srcLane) {
  // wave32 lane gather; wraps mod 32 (out-of-range sources masked by caller)
  return __int_as_float(
      __builtin_amdgcn_ds_bpermute(srcLane << 2, __float_as_int(v)));
}

static __device__ __forceinline__ float sigmoid_alpha(
    const float* __restrict__ la) {
  return 1.0f / (1.0f + expf(-la[0]));
}

static __device__ __forceinline__ float powi_f(float b, int e) {
  float r = 1.0f;
  while (e) { if (e & 1) r *= b; b *= b; e >>= 1; }
  return r;
}

// ---------------------------------------------------------------------------
// Kernel 1: per (row,chunk) compute A = sum_t c^{L-1-t} * (alpha*x[t]).
// One wave per chunk; each lane owns 4 interleaved streams (float4 loads),
// per-j ratio is c^128; final weights folded in, then butterfly reduce.
// Regular-temporal loads: this pass is what parks x in L2 for K3.
// ---------------------------------------------------------------------------
__global__ void ema_chunk_sum(const float* __restrict__ x,
                              const float* __restrict__ log_alpha,
                              float* __restrict__ Achunk) {
  const int lane  = threadIdx.x & 31;
  const int wave  = threadIdx.x >> 5;
  const int chunk = blockIdx.x * 8 + wave;       // 0..8191
  const int row   = chunk >> 4;
  const int cj    = chunk & (NCHUNK - 1);

  const float alpha = sigmoid_alpha(log_alpha);
  const float c = 1.0f - alpha;
  const float c2 = c * c, c4 = c2 * c2, c8 = c4 * c4, c16 = c8 * c8;
  const float c32 = c16 * c16, c64 = c32 * c32, c128 = c64 * c64;

  const v4f* xb =
      reinterpret_cast<const v4f*>(x + (size_t)row * T_LEN +
                                   (size_t)cj * LCHUNK) + lane;
  float a0 = 0.f, a1 = 0.f, a2 = 0.f, a3 = 0.f;
#pragma unroll
  for (int j = 0; j < LCHUNK / 128; ++j) {       // 16 iterations
    v4f v = xb[j * 32];
    a0 = a0 * c128 + v.x;
    a1 = a1 * c128 + v.y;
    a2 = a2 * c128 + v.z;
    a3 = a3 * c128 + v.w;
  }
  // element t=128j+4l+e weight: c^{128(J-1-j)} * c^{3-e} * c^{4(31-l)}
  float inner = ((a0 * c + a1) * c + a2) * c + a3;
  float part  = alpha * inner * powi_f(c4, 31 - lane);
#pragma unroll
  for (int off = 16; off >= 1; off >>= 1)
    part += shfl_f(part, lane ^ off);
  if (lane == 0) Achunk[chunk] = part;
}

// ---------------------------------------------------------------------------
// Kernel 2: per-row carry chain: s_0 = x[row,0]; s_{j+1} = A_j + c^L * s_j.
// ---------------------------------------------------------------------------
__global__ void ema_carries(const float* __restrict__ x,
                            const float* __restrict__ log_alpha,
                            const float* __restrict__ Achunk,
                            float* __restrict__ sIn) {
  const int row = blockIdx.x * blockDim.x + threadIdx.x;
  if (row >= B_ROWS) return;
  const float alpha = sigmoid_alpha(log_alpha);
  const float c  = 1.0f - alpha;
  const float cL = powf(c, (float)LCHUNK);
  float s = x[(size_t)row * T_LEN];              // init state = x[:,0]
#pragma unroll
  for (int j = 0; j < NCHUNK; ++j) {
    sIn[row * NCHUNK + j] = s;
    s = Achunk[row * NCHUNK + j] + cL * s;
  }
}

// ---------------------------------------------------------------------------
// Kernel 3: per-chunk scan with carry-in. One wave per chunk, 8 tiles of 256.
// Fully unrolled so all 16 global_load_b128 can issue ahead of the serial
// WMMA -> bpermute-scan -> readlane chain (loads don't depend on state s).
// ---------------------------------------------------------------------------
__global__ void ema_scan_wmma(const float* __restrict__ x,
                              const float* __restrict__ log_alpha,
                              const float* __restrict__ sIn,
                              float* __restrict__ y) {
  const int  lane  = threadIdx.x & 31;
  const int  wave  = threadIdx.x >> 5;
  const int  chunk = blockIdx.x * 8 + wave;
  const int  row   = chunk >> 4;
  const int  cj    = chunk & (NCHUNK - 1);
  const int  n     = lane & 15;
  const bool hig   = lane >= 16;

  const float alpha = sigmoid_alpha(log_alpha);
  const float c = 1.0f - alpha;

  float cp[16];                                  // c^0 .. c^15
  cp[0] = 1.0f;
#pragma unroll
  for (int i = 1; i < 16; ++i) cp[i] = cp[i - 1] * c;
  const float r  = cp[15] * c;                   // c^16 (segment ratio)
  const float rn = powi_f(r, n);                 // r^n per lane/column

  // A matrix (f16 16x32): L[m][k] = c^{m-k} for k<=m (k<16), else 0.
  // Lanes 0-15: row m=lane, halves 0-7 = K 0-7; lanes 16-31: row m=lane-16,
  // halves 0-7 = K 8-15; halves 8-15 = K 16..31 -> zero.
  v16h aMat;
#pragma unroll
  for (int h = 0; h < 16; ++h) {
    int   k = (h < 8) ? (hig ? h + 8 : h) : 100;  // h>=8 -> K>=16 -> zero
    float v = (k <= n) ? cp[n - k] : 0.0f;
    aMat[h] = (_Float16)v;
  }

  float s = sIn[row * NCHUNK + cj];              // chunk carry-in state
  const size_t base0   = (size_t)row * T_LEN + (size_t)cj * LCHUNK;
  const int    laneOff = n * 16 + (hig ? 8 : 0); // 8 consecutive floats/lane

#pragma unroll
  for (int t = 0; t < NTILE; ++t) {
    const v4f* px =
        reinterpret_cast<const v4f*>(x + base0 + t * TILE + laneOff);
    // Last use of x: nontemporal so the read stream doesn't fight for L2.
    v4f f0 = __builtin_nontemporal_load(px);
    v4f f1 = __builtin_nontemporal_load(px + 1);

    v16h bMat;                                   // B (f16 32x16), K>=16 zero
    bMat[0] = (_Float16)(alpha * f0.x);
    bMat[1] = (_Float16)(alpha * f0.y);
    bMat[2] = (_Float16)(alpha * f0.z);
    bMat[3] = (_Float16)(alpha * f0.w);
    bMat[4] = (_Float16)(alpha * f1.x);
    bMat[5] = (_Float16)(alpha * f1.y);
    bMat[6] = (_Float16)(alpha * f1.z);
    bMat[7] = (_Float16)(alpha * f1.w);
#pragma unroll
    for (int h = 8; h < 16; ++h) bMat[h] = (_Float16)0.0f;

    v8f acc = {};
    acc = __builtin_amdgcn_wmma_f32_16x16x32_f16(
        false, aMat, false, bMat, (short)0, acc, false, false);

    // Segment sums A_n = Y1[15][n] live in VGPR7 of lanes 16..31.
    float aN   = shfl_f(acc[7], 16 + n);         // broadcast to both halves
    float incl = aN;                             // inclusive scan, ratio r
    float rr   = r;
#pragma unroll
    for (int off = 1; off < 16; off <<= 1) {
      float up = shfl_f(incl, lane - off);
      if (n >= off) incl += rr * up;
      rr *= rr;
    }
    float ex    = shfl_f(incl, lane - 1);
    float carry = ((n == 0) ? 0.0f : ex) + rn * s;

    // Rank-1 update: y[m][n] = Y1[m][n] + c^{m+1} * carry_n
    float fac = hig ? cp[9] : cp[1];             // c^{m+1}, m = v + (hig?8:0)
    float yv[8];
#pragma unroll
    for (int v = 0; v < 8; ++v) {
      yv[v] = acc[v] + fac * carry;
      fac *= c;
    }
    s = shfl_f(yv[7], 31);                       // last element -> next tile

    v4f* py = reinterpret_cast<v4f*>(y + base0 + t * TILE + laneOff);
    v4f o0 = {yv[0], yv[1], yv[2], yv[3]};
    v4f o1 = {yv[4], yv[5], yv[6], yv[7]};
    // y is write-once, never re-read: nontemporal stores keep L2 for x.
    __builtin_nontemporal_store(o0, py);
    __builtin_nontemporal_store(o1, py + 1);
  }
}

// ---------------------------------------------------------------------------
extern "C" void kernel_launch(void* const* d_in, const int* in_sizes, int n_in,
                              void* d_out, int out_size, void* d_ws,
                              size_t ws_size, hipStream_t stream) {
  const float* x  = (const float*)d_in[0];
  const float* la = (const float*)d_in[1];
  float* yout     = (float*)d_out;

  float* Achunk = (float*)d_ws;                  // 512*16 floats
  float* sIn    = Achunk + B_ROWS * NCHUNK;      // 512*16 floats (64 KB total)

  const int nChunks = B_ROWS * NCHUNK;           // 8192 waves of work
  ema_chunk_sum<<<nChunks / 8, 256, 0, stream>>>(x, la, Achunk);
  ema_carries<<<(B_ROWS + 255) / 256, 256, 0, stream>>>(x, la, Achunk, sIn);
  ema_scan_wmma<<<nChunks / 8, 256, 0, stream>>>(x, la, sIn, yout);
}
